// Discriminator_27238682591239
// MI455X (gfx1250) — matrix operand order repaired and checked by reading, not compiled, not run on hardware
//
#include <hip/hip_runtime.h>
#include <hip/hip_bf16.h>
#include <math.h>

typedef _Float16 f16;
typedef _Float16 v16h __attribute__((ext_vector_type(16)));
typedef _Float16 v8h  __attribute__((ext_vector_type(8)));
typedef float    v8f  __attribute__((ext_vector_type(8)));

#define N_NODES 50000
#define N_EDGES 800000
#define N_GRAPH 512
#define F_IN    64
#define H_DIM   128
#define SEQ_DIM 256

// ---------------- utility kernels ----------------

__global__ void k_fill_f32(float* __restrict__ p, float v, int n) {
    int i = blockIdx.x * blockDim.x + threadIdx.x;
    if (i < n) p[i] = v;
}

__global__ void k_f32_to_f16(const float* __restrict__ in, f16* __restrict__ out, int n) {
    int i = blockIdx.x * blockDim.x + threadIdx.x;
    if (i < n) out[i] = (f16)in[i];
}

// W is [K, Nout] row-major (fan_in x fan_out). Produce f16 transpose [Nout, K].
__global__ void k_w_to_f16_t(const float* __restrict__ W, f16* __restrict__ Bt, int K, int Nout) {
    int i = blockIdx.x * blockDim.x + threadIdx.x;
    if (i < K * Nout) {
        int kk = i / Nout, nn = i % Nout;
        Bt[(size_t)nn * K + kk] = (f16)W[i];
    }
}

// ---------------- WMMA GEMM: out[M,128] = A[M,K](f16) @ B[K,128] + bias (+ Cin) ----------------
// One wave per 16x16 output tile. Bt is B transposed: [128, K] f16.
__global__ __launch_bounds__(32)
void k_gemm_wmma(const f16* __restrict__ A, const f16* __restrict__ Bt,
                 const float* __restrict__ bias, const float* __restrict__ Cin,
                 float* __restrict__ out, int K, int acc_in) {
    const int lane    = threadIdx.x;      // 0..31, all active
    const int rowbase = blockIdx.x * 16;  // M tile
    const int colbase = blockIdx.y * 16;  // N tile
    const int lh = lane & 15;
    const int hi = lane >> 4;

    v8f acc;
    if (acc_in) {
        const float* Crow = Cin + (size_t)(rowbase + hi * 8) * H_DIM + colbase + lh;
#pragma unroll
        for (int j = 0; j < 8; ++j) acc[j] = Crow[(size_t)j * H_DIM];
    } else {
#pragma unroll
        for (int j = 0; j < 8; ++j) acc[j] = 0.0f;
    }

    // A-frag: lane lh holds row (rowbase+lh); half i -> K = (i&7) + 8*hi + 16*(i>=8)
    const f16* Abase = A  + (size_t)(rowbase + lh) * K + hi * 8;
    // B-frag: lane lh holds col (colbase+lh); half i -> K = (i&7) + 8*hi + 16*(i>=8)
    const f16* Bbase = Bt + (size_t)(colbase + lh) * K + hi * 8;

    for (int kb = 0; kb < K; kb += 32) {
        v8h alo = *(const v8h*)(Abase + kb);
        v8h ahi = *(const v8h*)(Abase + kb + 16);
        v8h blo = *(const v8h*)(Bbase + kb);
        v8h bhi = *(const v8h*)(Bbase + kb + 16);
        v16h a, b;
#pragma unroll
        for (int i = 0; i < 8; ++i) {
            a[i] = alo[i]; a[i + 8] = ahi[i];
            b[i] = blo[i]; b[i + 8] = bhi[i];
        }
        acc = __builtin_amdgcn_wmma_f32_16x16x32_f16(
            /*neg_a=*/false, a, /*neg_b=*/false, b,
            /*c_mod=*/(short)0, acc, /*reuse_a=*/false, /*reuse_b=*/false);
    }

    const float bv = bias[colbase + lh];
    float* Drow = out + (size_t)(rowbase + hi * 8) * H_DIM + colbase + lh;
#pragma unroll
    for (int j = 0; j < 8; ++j) Drow[(size_t)j * H_DIM] = acc[j] + bv;
}

// ---------------- edge attention kernels ----------------

__device__ inline void atomicMaxF(float* addr, float val) {
    int* ia = (int*)addr;
    int old = *ia;
    while (__int_as_float(old) < val) {
        int assumed = old;
        old = atomicCAS(ia, assumed, __float_as_int(val));
        if (old == assumed) break;
    }
}

// one wave per edge: score = dot(q[dst], k[src]) / sqrt(128); atomicMax into m[dst]
__global__ void k_edge_score(const long long* __restrict__ src, const long long* __restrict__ dst,
                             const float* __restrict__ q, const float* __restrict__ kk,
                             float* __restrict__ score, float* __restrict__ m, int E) {
    int e    = blockIdx.x * 8 + (threadIdx.x >> 5);
    int lane = threadIdx.x & 31;
    if (e >= E) return;
    long long s = src[e], d = dst[e];
    float4 qa = ((const float4*)(q  + (size_t)d * H_DIM))[lane];
    float4 kb = ((const float4*)(kk + (size_t)s * H_DIM))[lane];
    float p = qa.x * kb.x + qa.y * kb.y + qa.z * kb.z + qa.w * kb.w;
#pragma unroll
    for (int off = 16; off > 0; off >>= 1) p += __shfl_xor(p, off, 32);
    if (lane == 0) {
        float sc = p * 0.08838834764831845f;  // 1/sqrt(128)
        score[e] = sc;
        atomicMaxF(&m[d], sc);
    }
}

__global__ void k_edge_exp(const long long* __restrict__ dst, const float* __restrict__ score,
                           const float* __restrict__ m, float* __restrict__ eexp,
                           float* __restrict__ denom, int E) {
    int e = blockIdx.x * blockDim.x + threadIdx.x;
    if (e >= E) return;
    long long d = dst[e];
    float ex = expf(score[e] - m[d]);
    eexp[e] = ex;
    atomicAdd(&denom[d], ex);
}

// one wave per edge: agg[dst] += alpha * v[src]
__global__ void k_edge_agg(const long long* __restrict__ src, const long long* __restrict__ dst,
                           const float* __restrict__ eexp, const float* __restrict__ denom,
                           const float* __restrict__ v, float* __restrict__ agg, int E) {
    int e    = blockIdx.x * 8 + (threadIdx.x >> 5);
    int lane = threadIdx.x & 31;
    if (e >= E) return;
    long long s = src[e], d = dst[e];
    float alpha = eexp[e] / denom[d];
    float4 vv = ((const float4*)(v + (size_t)s * H_DIM))[lane];
    float* ap = agg + (size_t)d * H_DIM + lane * 4;
    atomicAdd(ap + 0, alpha * vv.x);
    atomicAdd(ap + 1, alpha * vv.y);
    atomicAdd(ap + 2, alpha * vv.z);
    atomicAdd(ap + 3, alpha * vv.w);
}

// ---------------- activations / pooling / head ----------------

__global__ void k_relu_to_f16(const float* __restrict__ in, f16* __restrict__ out, int n) {
    int i = blockIdx.x * blockDim.x + threadIdx.x;
    if (i < n) out[i] = (f16)fmaxf(in[i], 0.0f);
}

__global__ void k_relu_f32(const float* __restrict__ in, float* __restrict__ out, int n) {
    int i = blockIdx.x * blockDim.x + threadIdx.x;
    if (i < n) out[i] = fmaxf(in[i], 0.0f);
}

// one wave per node: sums[batch[i]] += h[i]; cnt[batch[i]] += 1
__global__ void k_pool(const long long* __restrict__ batch, const float* __restrict__ h,
                       float* __restrict__ sums, float* __restrict__ cnt, int n) {
    int i    = blockIdx.x * 8 + (threadIdx.x >> 5);
    int lane = threadIdx.x & 31;
    if (i >= n) return;
    long long b = batch[i];
    float4 hv = ((const float4*)(h + (size_t)i * H_DIM))[lane];
    float* sp = sums + (size_t)b * H_DIM + lane * 4;
    atomicAdd(sp + 0, hv.x);
    atomicAdd(sp + 1, hv.y);
    atomicAdd(sp + 2, hv.z);
    atomicAdd(sp + 3, hv.w);
    if (lane == 0) atomicAdd(&cnt[b], 1.0f);
}

// seqr[g,h] = relu(seq[g] @ seq_w[:,h] + seq_b[h]); grid=G, block=H
__global__ void k_seq(const float* __restrict__ seqc, const float* __restrict__ w,
                      const float* __restrict__ b, float* __restrict__ outr) {
    int g = blockIdx.x, h = threadIdx.x;
    float acc = b[h];
    const float* sg = seqc + (size_t)g * SEQ_DIM;
    for (int j = 0; j < SEQ_DIM; ++j) acc = fmaf(sg[j], w[(size_t)j * H_DIM + h], acc);
    outr[(size_t)g * H_DIM + h] = fmaxf(acc, 0.0f);
}

// z[g,h] = relu(concat(pooled, seqr)[g] @ fc1_w[:,h] + fc1_b[h]); grid=G, block=H
__global__ void k_fc1(const float* __restrict__ sums, const float* __restrict__ cnt,
                      const float* __restrict__ seqr, const float* __restrict__ w,
                      const float* __restrict__ b, float* __restrict__ z) {
    int g = blockIdx.x, h = threadIdx.x;
    float inv = 1.0f / fmaxf(cnt[g], 1.0f);
    float acc = b[h];
    const float* sg = sums + (size_t)g * H_DIM;
    const float* qg = seqr + (size_t)g * H_DIM;
    for (int j = 0; j < H_DIM; ++j) acc = fmaf(sg[j] * inv, w[(size_t)j * H_DIM + h], acc);
    for (int j = 0; j < H_DIM; ++j) acc = fmaf(qg[j], w[(size_t)(H_DIM + j) * H_DIM + h], acc);
    z[(size_t)g * H_DIM + h] = fmaxf(acc, 0.0f);
}

// out[g] = sigmoid(z[g] @ fc2_w + fc2_b); one wave per graph
__global__ void k_fc2(const float* __restrict__ z, const float* __restrict__ w,
                      const float* __restrict__ b, float* __restrict__ out) {
    int g    = blockIdx.x * 8 + (threadIdx.x >> 5);
    int lane = threadIdx.x & 31;
    if (g >= N_GRAPH) return;
    float4 zv = ((const float4*)(z + (size_t)g * H_DIM))[lane];
    float4 wv = ((const float4*)w)[lane];
    float p = zv.x * wv.x + zv.y * wv.y + zv.z * wv.z + zv.w * wv.w;
#pragma unroll
    for (int off = 16; off > 0; off >>= 1) p += __shfl_xor(p, off, 32);
    if (lane == 0) out[g] = 1.0f / (1.0f + expf(-(p + b[0])));
}

// ---------------- launcher ----------------

extern "C" void kernel_launch(void* const* d_in, const int* in_sizes, int n_in,
                              void* d_out, int out_size, void* d_ws, size_t ws_size,
                              hipStream_t stream) {
    const float*     x     = (const float*)d_in[0];
    const long long* ei    = (const long long*)d_in[1];   // [2,E] int64
    const long long* batch = (const long long*)d_in[2];   // [N] int64
    const float*     seqc  = (const float*)d_in[3];
    const float *q1w=(const float*)d_in[4],  *q1b=(const float*)d_in[5];
    const float *k1w=(const float*)d_in[6],  *k1b=(const float*)d_in[7];
    const float *v1w=(const float*)d_in[8],  *v1b=(const float*)d_in[9];
    const float *s1w=(const float*)d_in[10], *s1b=(const float*)d_in[11];
    const float *q2w=(const float*)d_in[12], *q2b=(const float*)d_in[13];
    const float *k2w=(const float*)d_in[14], *k2b=(const float*)d_in[15];
    const float *v2w=(const float*)d_in[16], *v2b=(const float*)d_in[17];
    const float *s2w=(const float*)d_in[18], *s2b=(const float*)d_in[19];
    const float *seqw=(const float*)d_in[20], *seqb=(const float*)d_in[21];
    const float *f1w=(const float*)d_in[22], *f1b=(const float*)d_in[23];
    const float *f2w=(const float*)d_in[24], *f2b=(const float*)d_in[25];
    float* out = (float*)d_out;

    const long long* esrc = ei;
    const long long* edst = ei + N_EDGES;

    // workspace carve-out (256B aligned)
    char* wp = (char*)d_ws;
    auto carve = [&](size_t bytes) -> char* {
        char* r = wp;
        wp += (bytes + 255) & ~(size_t)255;
        return r;
    };
    f16*   xh    = (f16*)  carve((size_t)N_NODES * F_IN * sizeof(f16));
    f16*   q1t   = (f16*)  carve((size_t)F_IN * H_DIM * sizeof(f16));
    f16*   k1t   = (f16*)  carve((size_t)F_IN * H_DIM * sizeof(f16));
    f16*   v1t   = (f16*)  carve((size_t)F_IN * H_DIM * sizeof(f16));
    f16*   s1t   = (f16*)  carve((size_t)F_IN * H_DIM * sizeof(f16));
    f16*   q2t   = (f16*)  carve((size_t)H_DIM * H_DIM * sizeof(f16));
    f16*   k2t   = (f16*)  carve((size_t)H_DIM * H_DIM * sizeof(f16));
    f16*   v2t   = (f16*)  carve((size_t)H_DIM * H_DIM * sizeof(f16));
    f16*   s2t   = (f16*)  carve((size_t)H_DIM * H_DIM * sizeof(f16));
    float* qb    = (float*)carve((size_t)N_NODES * H_DIM * sizeof(float));
    float* kb    = (float*)carve((size_t)N_NODES * H_DIM * sizeof(float));
    float* vb    = (float*)carve((size_t)N_NODES * H_DIM * sizeof(float));
    float* agg   = (float*)carve((size_t)N_NODES * H_DIM * sizeof(float));
    f16*   h1h   = (f16*)  carve((size_t)N_NODES * H_DIM * sizeof(f16));
    float* score = (float*)carve((size_t)N_EDGES * sizeof(float));
    float* eexp  = (float*)carve((size_t)N_EDGES * sizeof(float));
    float* mmax  = (float*)carve((size_t)N_NODES * sizeof(float));
    float* denom = (float*)carve((size_t)N_NODES * sizeof(float));
    float* sums  = (float*)carve((size_t)N_GRAPH * H_DIM * sizeof(float));
    float* cnt   = (float*)carve((size_t)N_GRAPH * sizeof(float));
    float* seqr  = (float*)carve((size_t)N_GRAPH * H_DIM * sizeof(float));
    float* z2    = (float*)carve((size_t)N_GRAPH * H_DIM * sizeof(float));

    const int T = 256;
    auto cdiv = [](int a, int b) { return (a + b - 1) / b; };

    // ---- precision conversion ----
    k_f32_to_f16<<<cdiv(N_NODES * F_IN, T), T, 0, stream>>>(x, xh, N_NODES * F_IN);
    k_w_to_f16_t<<<cdiv(F_IN * H_DIM, T), T, 0, stream>>>(q1w, q1t, F_IN, H_DIM);
    k_w_to_f16_t<<<cdiv(F_IN * H_DIM, T), T, 0, stream>>>(k1w, k1t, F_IN, H_DIM);
    k_w_to_f16_t<<<cdiv(F_IN * H_DIM, T), T, 0, stream>>>(v1w, v1t, F_IN, H_DIM);
    k_w_to_f16_t<<<cdiv(F_IN * H_DIM, T), T, 0, stream>>>(s1w, s1t, F_IN, H_DIM);
    k_w_to_f16_t<<<cdiv(H_DIM * H_DIM, T), T, 0, stream>>>(q2w, q2t, H_DIM, H_DIM);
    k_w_to_f16_t<<<cdiv(H_DIM * H_DIM, T), T, 0, stream>>>(k2w, k2t, H_DIM, H_DIM);
    k_w_to_f16_t<<<cdiv(H_DIM * H_DIM, T), T, 0, stream>>>(v2w, v2t, H_DIM, H_DIM);
    k_w_to_f16_t<<<cdiv(H_DIM * H_DIM, T), T, 0, stream>>>(s2w, s2t, H_DIM, H_DIM);

    const dim3 gemmGrid(N_NODES / 16, H_DIM / 16);
    const int  nh  = N_NODES * H_DIM;
    const int  eb8 = cdiv(N_EDGES, 8);

    // ---- shared per-layer pipeline ----
    auto layer = [&](const f16* Ah, int K,
                     const f16* qt, const float* qbi, const f16* kt, const float* kbi,
                     const f16* vt, const float* vbi, const f16* st, const float* sbi) {
        k_gemm_wmma<<<gemmGrid, 32, 0, stream>>>(Ah, qt, qbi, nullptr, qb, K, 0);
        k_gemm_wmma<<<gemmGrid, 32, 0, stream>>>(Ah, kt, kbi, nullptr, kb, K, 0);
        k_gemm_wmma<<<gemmGrid, 32, 0, stream>>>(Ah, vt, vbi, nullptr, vb, K, 0);
        k_fill_f32<<<cdiv(N_NODES, T), T, 0, stream>>>(mmax, -INFINITY, N_NODES);
        k_fill_f32<<<cdiv(N_NODES, T), T, 0, stream>>>(denom, 0.0f, N_NODES);
        k_fill_f32<<<cdiv(nh, T), T, 0, stream>>>(agg, 0.0f, nh);
        k_edge_score<<<eb8, T, 0, stream>>>(esrc, edst, qb, kb, score, mmax, N_EDGES);
        k_edge_exp<<<cdiv(N_EDGES, T), T, 0, stream>>>(edst, score, mmax, eexp, denom, N_EDGES);
        k_edge_agg<<<eb8, T, 0, stream>>>(esrc, edst, eexp, denom, vb, agg, N_EDGES);
        // fused skip: agg = agg + Ah @ sw + sb   (accumulate mode)
        k_gemm_wmma<<<gemmGrid, 32, 0, stream>>>(Ah, st, sbi, agg, agg, K, 1);
    };

    // ---- layer 1 ----
    layer(xh, F_IN, q1t, q1b, k1t, k1b, v1t, v1b, s1t, s1b);
    k_relu_to_f16<<<cdiv(nh, T), T, 0, stream>>>(agg, h1h, nh);   // h1 (f16) feeds layer 2

    // ---- layer 2 ----
    layer(h1h, H_DIM, q2t, q2b, k2t, k2b, v2t, v2b, s2t, s2b);
    k_relu_f32<<<cdiv(nh, T), T, 0, stream>>>(agg, qb, nh);       // h2 (f32) reuses q buffer

    // ---- global mean pool ----
    k_fill_f32<<<cdiv(N_GRAPH * H_DIM, T), T, 0, stream>>>(sums, 0.0f, N_GRAPH * H_DIM);
    k_fill_f32<<<cdiv(N_GRAPH, T), T, 0, stream>>>(cnt, 0.0f, N_GRAPH);
    k_pool<<<cdiv(N_NODES, 8), T, 0, stream>>>(batch, qb, sums, cnt, N_NODES);

    // ---- head ----
    k_seq<<<N_GRAPH, H_DIM, 0, stream>>>(seqc, seqw, seqb, seqr);
    k_fc1<<<N_GRAPH, H_DIM, 0, stream>>>(sums, cnt, seqr, f1w, f1b, z2);
    k_fc2<<<cdiv(N_GRAPH, 8), T, 0, stream>>>(z2, f2w, f2b, out);
}
